// PDGGNN_3023656976525
// MI455X (gfx1250) — matrix operand order repaired and checked by reading, not compiled, not run on hardware
//
#include <hip/hip_runtime.h>
#include <hip/hip_bf16.h>
#include <stdint.h>

#define N_NODES 50000
#define N_EDGES 800000
#define DIN     128
#define HD      96
#define KEXP    8
#define NLAYERS 4
#define COUT    40
#define ROWTILES (N_NODES / 16)   // 3125, exact

// LDS row strides chosen so stride-in-dwords is odd (49, 65) -> conflict-free ds_load_b32
#define LDSH (HD + 2)    // 98 ushort = 49 dwords
#define LDSX (DIN + 2)   // 130 ushort = 65 dwords

typedef __attribute__((ext_vector_type(16))) __bf16 v16bf;
typedef __attribute__((ext_vector_type(8)))  float  v8f;

union Frag16 { unsigned int u[8]; v16bf v; };

__device__ __forceinline__ unsigned short f2bf(float f) {
    unsigned int u = __float_as_uint(f);
    u += 0x7FFFu + ((u >> 16) & 1u);           // round-to-nearest-even
    return (unsigned short)(u >> 16);
}

// A fragment (16x32 bf16) from LDS, row-major with row stride `stride` (ushorts).
// Layout per ISA 7.12.2: lane m = l&15, half = l>>4; VGPR vg holds K pair
// k = (vg>>2)*16 + half*8 + (vg&3)*2.
__device__ __forceinline__ v16bf load_a_lds(const unsigned short* base, int stride,
                                            int k0, int lane) {
    Frag16 f;
    const unsigned short* row = base + (lane & 15) * stride + k0;
    int kh = (lane >> 4) * 8;
#pragma unroll
    for (int vg = 0; vg < 8; ++vg) {
        int k = ((vg >> 2) << 4) + kh + ((vg & 3) << 1);
        f.u[vg] = *(const unsigned int*)(row + k);
    }
    return f.v;
}

// B fragment (32x16 bf16) from global pre-transposed weights Wt[n][k] (row stride strideK).
// Layout: lane n = n0 + (l&15); half = l>>4; VGPR vg holds K pair k = (vg>>2)*16 + half*8 + (vg&3)*2.
__device__ __forceinline__ v16bf load_b_gl(const unsigned short* __restrict__ wt,
                                           int strideK, int n0, int k0, int lane) {
    Frag16 f;
    const unsigned short* p = wt + (n0 + (lane & 15)) * strideK + k0 + ((lane >> 4) << 3);
#pragma unroll
    for (int vg = 0; vg < 8; ++vg)
        f.u[vg] = *(const unsigned int*)(p + ((vg >> 2) << 4) + ((vg & 3) << 1));
    return f.v;
}

// ---------------- weight prep: fp32 -> bf16 + transpose to [n][k] ----------------
__global__ void prep_kernel(const float* __restrict__ fc0_w,
                            const float* __restrict__ fc1_w,
                            const float* __restrict__ conv_w,
                            unsigned short* __restrict__ fc0_wt,
                            unsigned short* __restrict__ conv_wt,
                            unsigned short* __restrict__ fc1_wt) {
    int t = blockIdx.x * blockDim.x + threadIdx.x;
    const int nconv = NLAYERS * KEXP * HD * HD;          // 294912
    if (t < nconv) {
        int o  = t % HD;
        int d  = (t / HD) % HD;
        int lk = t / (HD * HD);
        conv_wt[(lk * HD + o) * HD + d] = f2bf(conv_w[(lk * HD + d) * HD + o]);
    }
    if (t < DIN * HD) {                                   // fc0_w [DIN][HD] -> [HD][DIN]
        int k = t / HD, n = t % HD;
        fc0_wt[n * DIN + k] = f2bf(fc0_w[k * HD + n]);
    }
    if (t < 48 * HD) {                                    // fc1_w [HD][COUT] -> [48][HD], zero pad
        int o = t / HD, d = t % HD;
        fc1_wt[o * HD + d] = (o < COUT) ? f2bf(fc1_w[d * COUT + o]) : (unsigned short)0;
    }
}

// ---------------- fc0: h = relu(x @ W0 + b0), WMMA bf16 ----------------
__global__ __launch_bounds__(192) void fc0_kernel(const float* __restrict__ x,
                                                  const unsigned short* __restrict__ wt,
                                                  const float* __restrict__ bias,
                                                  float* __restrict__ h) {
    __shared__ unsigned short Ax[16 * LDSX];
    const int row0 = blockIdx.x * 16;
    const int tid  = threadIdx.x;
    for (int i = tid; i < 16 * DIN; i += 192) {
        int m = i / DIN, k = i % DIN;
        Ax[m * LDSX + k] = f2bf(x[(row0 + m) * DIN + k]);
    }
    __syncthreads();
    const int lane = tid & 31;
    const int n0   = (tid >> 5) * 16;                    // 6 waves -> 6 col tiles of 96
    const int n    = n0 + (lane & 15);
    const int mh   = (lane >> 4) * 8;
    v8f acc;
    float b = bias[n];
#pragma unroll
    for (int v = 0; v < 8; ++v) acc[v] = b;
#pragma unroll
    for (int kc = 0; kc < 4; ++kc) {
        v16bf a  = load_a_lds(Ax, LDSX, kc * 32, lane);
        v16bf bf = load_b_gl(wt, DIN, n0, kc * 32, lane);
        acc = __builtin_amdgcn_wmma_f32_16x16x32_bf16(false, a, false, bf,
                                                      (short)0, acc, false, false);
    }
#pragma unroll
    for (int v = 0; v < 8; ++v) {
        float r = acc[v];
        h[(row0 + mh + v) * HD + n] = r > 0.f ? r : 0.f;
    }
}

// ---------------- routing: z = softmax(h @ ctx_w + ctx_b) ----------------
__global__ void route_kernel(const float* __restrict__ h,
                             const float* __restrict__ cw,   // [HD][KEXP] for this layer
                             const float* __restrict__ cb,   // [KEXP]
                             float* __restrict__ z) {
    int n = blockIdx.x * blockDim.x + threadIdx.x;
    if (n >= N_NODES) return;
    float lg[KEXP];
#pragma unroll
    for (int k = 0; k < KEXP; ++k) lg[k] = cb[k];
    for (int d = 0; d < HD; ++d) {
        float hv = h[n * HD + d];
#pragma unroll
        for (int k = 0; k < KEXP; ++k) lg[k] += hv * cw[d * KEXP + k];
    }
    float mx = lg[0];
#pragma unroll
    for (int k = 1; k < KEXP; ++k) mx = fmaxf(mx, lg[k]);
    float s = 0.f;
#pragma unroll
    for (int k = 0; k < KEXP; ++k) { lg[k] = __expf(lg[k] - mx); s += lg[k]; }
    float inv = 1.f / s;
#pragma unroll
    for (int k = 0; k < KEXP; ++k) z[n * KEXP + k] = lg[k] * inv;
}

// ---------------- SpMM: hi[row] += h[col] over edges ----------------
__global__ void zero_kernel(float* __restrict__ p, int count) {
    int i = blockIdx.x * blockDim.x + threadIdx.x;
    if (i < count) p[i] = 0.f;
}

__global__ __launch_bounds__(96) void scatter_kernel(const float* __restrict__ h,
                                                     const long long* __restrict__ ei,
                                                     float* __restrict__ hi) {
    int e = blockIdx.x;
    int r = (int)ei[e];
    int c = (int)ei[N_EDGES + e];
    int f = threadIdx.x;
    atomicAdd(&hi[r * HD + f], h[c * HD + f]);
}

// ---------------- expert combine: h = relu(sum_k z_k * (hi @ W_k) + h) ----------------
__global__ __launch_bounds__(192) void expert_kernel(const float* __restrict__ hi,
                                                     const float* __restrict__ z,
                                                     const unsigned short* __restrict__ cwt, // [KEXP][HD][HD] this layer
                                                     float* __restrict__ h) {
    __shared__ unsigned short Hb[16 * LDSH];
    __shared__ float Zt[16][KEXP];
    const int row0 = blockIdx.x * 16;
    const int tid  = threadIdx.x;
    for (int i = tid; i < 16 * HD; i += 192) {
        int m = i / HD, k = i % HD;
        Hb[m * LDSH + k] = f2bf(hi[(row0 + m) * HD + k]);
    }
    if (tid < 16 * KEXP)
        Zt[tid >> 3][tid & 7] = z[(row0 + (tid >> 3)) * KEXP + (tid & 7)];
    __syncthreads();
    const int lane = tid & 31;
    const int n0   = (tid >> 5) * 16;
    const int n    = n0 + (lane & 15);
    const int mh   = (lane >> 4) * 8;
    float out[8];
#pragma unroll
    for (int v = 0; v < 8; ++v) out[v] = h[(row0 + mh + v) * HD + n];   // residual
    // A fragments reused across all 8 experts (hi tile fixed)
    v16bf a0 = load_a_lds(Hb, LDSH,  0, lane);
    v16bf a1 = load_a_lds(Hb, LDSH, 32, lane);
    v16bf a2 = load_a_lds(Hb, LDSH, 64, lane);
#pragma unroll
    for (int k = 0; k < KEXP; ++k) {
        const unsigned short* wk = cwt + k * HD * HD;
        v8f acc = {};
        acc = __builtin_amdgcn_wmma_f32_16x16x32_bf16(false, a0, false,
              load_b_gl(wk, HD, n0,  0, lane), (short)0, acc, false, false);
        acc = __builtin_amdgcn_wmma_f32_16x16x32_bf16(false, a1, false,
              load_b_gl(wk, HD, n0, 32, lane), (short)0, acc, false, false);
        acc = __builtin_amdgcn_wmma_f32_16x16x32_bf16(false, a2, false,
              load_b_gl(wk, HD, n0, 64, lane), (short)0, acc, false, false);
#pragma unroll
        for (int v = 0; v < 8; ++v) out[v] += Zt[mh + v][k] * acc[v];
    }
#pragma unroll
    for (int v = 0; v < 8; ++v) {
        float r = out[v];
        h[(row0 + mh + v) * HD + n] = r > 0.f ? r : 0.f;
    }
}

// ---------------- fc1: out = h @ W1 + b1 (40 cols, padded to 48) ----------------
__global__ __launch_bounds__(96) void fc1_kernel(const float* __restrict__ h,
                                                 const unsigned short* __restrict__ wt,
                                                 const float* __restrict__ bias,
                                                 float* __restrict__ out) {
    __shared__ unsigned short Hb[16 * LDSH];
    const int row0 = blockIdx.x * 16;
    const int tid  = threadIdx.x;
    for (int i = tid; i < 16 * HD; i += 96) {
        int m = i / HD, k = i % HD;
        Hb[m * LDSH + k] = f2bf(h[(row0 + m) * HD + k]);
    }
    __syncthreads();
    const int lane = tid & 31;
    const int n0   = (tid >> 5) * 16;                    // 3 waves -> 48 padded cols
    const int n    = n0 + (lane & 15);
    const int mh   = (lane >> 4) * 8;
    v8f acc;
    float b = (n < COUT) ? bias[n] : 0.f;
#pragma unroll
    for (int v = 0; v < 8; ++v) acc[v] = b;
#pragma unroll
    for (int kc = 0; kc < 3; ++kc) {
        v16bf a  = load_a_lds(Hb, LDSH, kc * 32, lane);
        v16bf bf = load_b_gl(wt, HD, n0, kc * 32, lane);
        acc = __builtin_amdgcn_wmma_f32_16x16x32_bf16(false, a, false, bf,
                                                      (short)0, acc, false, false);
    }
    if (n < COUT) {
#pragma unroll
        for (int v = 0; v < 8; ++v)
            out[(row0 + mh + v) * COUT + n] = acc[v];
    }
}

extern "C" void kernel_launch(void* const* d_in, const int* in_sizes, int n_in,
                              void* d_out, int out_size, void* d_ws, size_t ws_size,
                              hipStream_t stream) {
    const float*     x      = (const float*)d_in[0];
    const long long* ei     = (const long long*)d_in[1];   // int64 edge_index [2,E]
    const float*     fc0_w  = (const float*)d_in[2];
    const float*     fc0_b  = (const float*)d_in[3];
    const float*     fc1_w  = (const float*)d_in[4];
    const float*     fc1_b  = (const float*)d_in[5];
    const float*     ctx_w  = (const float*)d_in[6];       // [L][HD][KEXP]
    const float*     ctx_b  = (const float*)d_in[7];       // [L][KEXP]
    const float*     conv_w = (const float*)d_in[8];       // [L][KEXP][HD][HD]
    float* outp = (float*)d_out;

    // workspace layout (all offsets 256B-aligned)
    char* ws = (char*)d_ws;
    float*          h       = (float*)(ws + 0);                       // N*HD f32
    float*          hi      = (float*)(ws + 19200000);                // N*HD f32
    float*          z       = (float*)(ws + 38400000);                // N*KEXP f32
    unsigned short* fc0_wt  = (unsigned short*)(ws + 40000000);       // [HD][DIN] bf16
    unsigned short* conv_wt = (unsigned short*)(ws + 40024576);       // [L][K][HD][HD] bf16
    unsigned short* fc1_wt  = (unsigned short*)(ws + 40614400);       // [48][HD] bf16

    const int nconv = NLAYERS * KEXP * HD * HD;
    prep_kernel<<<(nconv + 255) / 256, 256, 0, stream>>>(fc0_w, fc1_w, conv_w,
                                                         fc0_wt, conv_wt, fc1_wt);

    fc0_kernel<<<ROWTILES, 192, 0, stream>>>(x, fc0_wt, fc0_b, h);

    for (int l = 0; l < NLAYERS; ++l) {
        route_kernel<<<(N_NODES + 255) / 256, 256, 0, stream>>>(
            h, ctx_w + (size_t)l * HD * KEXP, ctx_b + (size_t)l * KEXP, z);
        zero_kernel<<<(N_NODES * HD + 255) / 256, 256, 0, stream>>>(hi, N_NODES * HD);
        scatter_kernel<<<N_EDGES, 96, 0, stream>>>(h, ei, hi);
        expert_kernel<<<ROWTILES, 192, 0, stream>>>(
            hi, z, conv_wt + (size_t)l * KEXP * HD * HD, h);
    }

    fc1_kernel<<<ROWTILES, 96, 0, stream>>>(h, fc1_wt, fc1_b, outp);
}
